// MultiHeadAttention_91130616086535
// MI455X (gfx1250) — compile-verified, matrix operand order checked by reading, not run on hardware
//
#include <hip/hip_runtime.h>

// ---------------------------------------------------------------------------
// BERT attention block on gfx1250 (wave32, WMMA bf16 16x16x32).
//   H=768, NH=12, HD=64, B=2, S=2048
// - One-time f32->bf16 conversion; all GEMMs stream raw bf16 b128 fragments.
// - Flash attention stages K/V chunks into LDS once per block via
//   global_load_async_to_lds_b128 (ASYNCcnt), double-buffered.
// ---------------------------------------------------------------------------

constexpr int kH  = 768;
constexpr int kNH = 12;
constexpr int kHD = 64;
constexpr int kB  = 2;
constexpr int kS  = 2048;
constexpr int kM  = kB * kS;          // 4096 total rows
constexpr int kBH = kB * kNH;         // 24 (batch*heads)

constexpr int kKStride = 72;          // LDS row stride for K chunk (pad 8)
constexpr int kVStride = 40;          // LDS row stride for V chunk (pad 8)

typedef unsigned short u16;
typedef __attribute__((ext_vector_type(16))) __bf16 v16bf;
typedef __attribute__((ext_vector_type(16))) u16    v16us;
typedef __attribute__((ext_vector_type(8)))  u16    v8us;
typedef __attribute__((ext_vector_type(8)))  float  v8f;
typedef __attribute__((ext_vector_type(4)))  float  v4f;

typedef __attribute__((address_space(3))) u16 lds_u16;

__device__ __forceinline__ u16 f2bf_u(float f) {
  unsigned u = __builtin_bit_cast(unsigned, f);
  u += 0x7FFFu + ((u >> 16) & 1u);    // round-to-nearest-even
  return (u16)(u >> 16);
}

// Per-lane slice of a 16x32 bf16 WMMA A/B fragment from a row-major bf16
// matrix.  p points at element (row = lane&15, k = k0 + c0), c0 = 8*(lane>=16).
// Lane needs k-offsets {0..7} and {16..23}: two 16-byte loads, no ALU.
__device__ __forceinline__ v16bf load_frag_bf16(const u16* p) {
  v8us a = *(const v8us*)(p);
  v8us b = *(const v8us*)(p + 16);
  v16us u = __builtin_shufflevector(a, b, 0, 1, 2, 3, 4, 5, 6, 7,
                                          8, 9, 10, 11, 12, 13, 14, 15);
  return __builtin_bit_cast(v16bf, u);
}

__device__ __forceinline__ v8f wmma_bf16(v16bf a, v16bf b, v8f c) {
  return __builtin_amdgcn_wmma_f32_16x16x32_bf16(false, a, false, b,
                                                 (short)0, c, false, false);
}

// Async DMA of 16 bytes global -> LDS (tracked by ASYNCcnt).
__device__ __forceinline__ void async_copy_b128(u16* lds_dst, const u16* gsrc) {
  lds_u16* lp = (lds_u16*)lds_dst;
  asm volatile("global_load_async_to_lds_b128 %0, %1, off"
               :
               : "v"(lp), "v"(gsrc)
               : "memory");
}

__device__ __forceinline__ void wait_async0() {
  asm volatile("s_wait_asynccnt 0" ::: "memory");
}

// ---------------------------------------------------------------------------
// Kernel 0: elementwise f32 -> bf16 (8 elements / thread, vectorized).
// ---------------------------------------------------------------------------
__global__ void __launch_bounds__(256)
cvt_bf16_kernel(const float* __restrict__ in, u16* __restrict__ out, int n) {
  const int i = (blockIdx.x * 256 + threadIdx.x) * 8;
  if (i + 8 <= n) {
    v4f a = *(const v4f*)(in + i);
    v4f b = *(const v4f*)(in + i + 4);
    v8us u;
#pragma unroll
    for (int j = 0; j < 4; ++j) { u[j] = f2bf_u(a[j]); u[4 + j] = f2bf_u(b[j]); }
    *(v8us*)(out + i) = u;
  }
}

// ---------------------------------------------------------------------------
// Kernel 1: QKV projection from bf16 operands.  out = x @ W^T + b.
// Q,K written bf16 [BH, S, HD]; V written TRANSPOSED bf16 [BH, HD, S].
// ---------------------------------------------------------------------------
__global__ void __launch_bounds__(256)
qkv_proj_kernel(const u16* __restrict__ xbf,
                const u16* __restrict__ Wqb, const float* __restrict__ bq,
                const u16* __restrict__ Wkb, const float* __restrict__ bk,
                const u16* __restrict__ Wvb, const float* __restrict__ bv,
                u16* __restrict__ Qo, u16* __restrict__ Ko,
                u16* __restrict__ Vt) {
  const int lane = threadIdx.x & 31;
  const int wave = threadIdx.x >> 5;
  const int lh   = lane >> 4;
  const int ml   = lane & 15;
  const int c0   = lh * 8;
  const int m0   = (blockIdx.x * 8 + wave) * 16;
  const int n0   = blockIdx.y * 16;
  const int z    = blockIdx.z;

  const u16* W;  const float* bias;
  if (z == 0)      { W = Wqb; bias = bq; }
  else if (z == 1) { W = Wkb; bias = bk; }
  else             { W = Wvb; bias = bv; }

  v8f acc = {};
  const u16* arow = xbf + (size_t)(m0 + ml) * kH + c0;
  const u16* brow = W   + (size_t)(n0 + ml) * kH + c0;
#pragma unroll 4
  for (int k0 = 0; k0 < kH; k0 += 32)
    acc = wmma_bf16(load_frag_bf16(arow + k0), load_frag_bf16(brow + k0), acc);

  const int   n   = n0 + ml;
  const int   h   = n / kHD;
  const int   d   = n % kHD;
  const float bb_ = bias[n];
#pragma unroll
  for (int r = 0; r < 8; ++r) {
    const int m  = m0 + r + 8 * lh;
    const int bb = m / kS, ss = m % kS;
    const int bh = bb * kNH + h;
    const u16 v  = f2bf_u(acc[r] + bb_);
    if (z == 0)      Qo[((size_t)bh * kS + ss) * kHD + d] = v;
    else if (z == 1) Ko[((size_t)bh * kS + ss) * kHD + d] = v;
    else             Vt[((size_t)bh * kHD + d) * kS + ss] = v;
  }
}

// ---------------------------------------------------------------------------
// Kernel 2: flash attention.  8 waves / block share one (b,h); K/V chunks are
// DMA'd into double-buffered LDS with global_load_async_to_lds_b128, then each
// wave streams WMMA fragments from LDS.  Online softmax; P tile transposed
// C-layout -> A-layout through per-wave LDS.  Writes ctx bf16 [B,S,H].
// ---------------------------------------------------------------------------
__global__ void __launch_bounds__(256)
flash_attn_kernel(const u16* __restrict__ Q, const u16* __restrict__ K,
                  const u16* __restrict__ Vt,
                  const float* __restrict__ amask,
                  u16* __restrict__ ctx) {
  __shared__ __align__(16) u16 kbuf[2][32 * kKStride];  // 32 kv-rows x 64 dims
  __shared__ __align__(16) u16 vbuf[2][64 * kVStride];  // 64 d-rows  x 32 kv
  __shared__ float pbuf[8][16 * 33];                    // per-wave P tile

  const int tid  = threadIdx.x;
  const int lane = tid & 31;
  const int wave = tid >> 5;
  const int lh   = lane >> 4;
  const int ml   = lane & 15;
  const int c0   = lh * 8;

  const int bh = blockIdx.y;                  // 0..B*NH-1
  const int b  = bh / kNH;
  const int q0 = (blockIdx.x * 8 + wave) * 16;

  const u16* Qb  = Q  + (size_t)bh * kS * kHD;
  const u16* Kb  = K  + (size_t)bh * kS * kHD;
  const u16* Vtb = Vt + (size_t)bh * kHD * kS;

  // Staging coordinates: K chunk = 256 threads x 16B, V chunk likewise.
  const int krow = tid >> 3, kseg = tid & 7;  // 32 rows x 8 segs
  const int vrow = tid >> 2, vseg = tid & 3;  // 64 rows x 4 segs

  // Q fragments for this tile (HD=64 -> two k-steps of 32), loaded once.
  const v16bf qf0 = load_frag_bf16(Qb + (size_t)(q0 + ml) * kHD + 0  + c0);
  const v16bf qf1 = load_frag_bf16(Qb + (size_t)(q0 + ml) * kHD + 32 + c0);

  v8f o0 = {}, o1 = {}, o2 = {}, o3 = {};     // 16x64 output accumulator
  float mrun[8], lrun[8];
#pragma unroll
  for (int r = 0; r < 8; ++r) { mrun[r] = -3.0e38f; lrun[r] = 0.0f; }

  float* pw = &pbuf[wave][0];

  // Kick off DMA for chunk 0 into buffer 0.
  async_copy_b128(&kbuf[0][krow * kKStride + kseg * 8],
                  Kb + (size_t)krow * kHD + kseg * 8);
  async_copy_b128(&vbuf[0][vrow * kVStride + vseg * 8],
                  Vtb + (size_t)vrow * kS + vseg * 8);

  for (int kv0 = 0; kv0 < kS; kv0 += 32) {
    const int cur = (kv0 >> 5) & 1;
    wait_async0();                    // my chunk DMA done
    __syncthreads();                  // everyone's DMA done, prev buf free

    // Prefetch next chunk into the other buffer (overlaps with compute).
    if (kv0 + 32 < kS) {
      const int nxt = cur ^ 1;
      async_copy_b128(&kbuf[nxt][krow * kKStride + kseg * 8],
                      Kb + (size_t)(kv0 + 32 + krow) * kHD + kseg * 8);
      async_copy_b128(&vbuf[nxt][vrow * kVStride + vseg * 8],
                      Vtb + (size_t)vrow * kS + (kv0 + 32) + vseg * 8);
    }

    const u16* kch = &kbuf[cur][0];
    const u16* vch = &vbuf[cur][0];

    // ---- scores: two 16x16 tiles covering kv columns [kv0, kv0+32) ----
    v8f s0, s1;
#pragma unroll
    for (int t = 0; t < 2; ++t) {
      v8f sc = {};
      const u16* krow_l = kch + (t * 16 + ml) * kKStride + c0;
      sc = wmma_bf16(qf0, load_frag_bf16(krow_l + 0),  sc);
      sc = wmma_bf16(qf1, load_frag_bf16(krow_l + 32), sc);
      const float mval =
          (1.0f - amask[(size_t)b * kS + kv0 + t * 16 + ml]) * -10000.0f;
      v8f sv;
#pragma unroll
      for (int r = 0; r < 8; ++r) sv[r] = sc[r] * 0.125f + mval;  // 1/sqrt(64)
      if (t == 0) s0 = sv; else s1 = sv;
    }

    // ---- online softmax: row max / rescale / exp / row sum ----
#pragma unroll
    for (int r = 0; r < 8; ++r) {
      float mx = fmaxf(s0[r], s1[r]);
#pragma unroll
      for (int off = 1; off < 16; off <<= 1)
        mx = fmaxf(mx, __shfl_xor(mx, off, 32));
      const float mnew  = fmaxf(mrun[r], mx);
      const float alpha = __expf(mrun[r] - mnew);
      const float p0 = __expf(s0[r] - mnew);
      const float p1 = __expf(s1[r] - mnew);
      s0[r] = p0; s1[r] = p1;
      float rs = p0 + p1;
#pragma unroll
      for (int off = 1; off < 16; off <<= 1)
        rs += __shfl_xor(rs, off, 32);
      lrun[r] = lrun[r] * alpha + rs;
      mrun[r] = mnew;
      o0[r] *= alpha; o1[r] *= alpha; o2[r] *= alpha; o3[r] *= alpha;
    }

    // ---- transpose P (C-layout) -> A-layout via per-wave LDS ----
#pragma unroll
    for (int r = 0; r < 8; ++r) {
      const int row = r + 8 * lh;
      pw[row * 33 + ml]      = s0[r];
      pw[row * 33 + 16 + ml] = s1[r];
    }
    // Same-wave LDS ops are in-order; compiler inserts s_wait_dscnt for deps.
    v16us pu;
    const float* prow = pw + ml * 33 + c0;
#pragma unroll
    for (int i = 0; i < 8; ++i) pu[i]     = f2bf_u(prow[i]);
#pragma unroll
    for (int i = 0; i < 8; ++i) pu[8 + i] = f2bf_u(prow[16 + i]);
    const v16bf pf = __builtin_bit_cast(v16bf, pu);

    // ---- P x V : four 16-wide d tiles across HD=64 (LDS b128 loads) ----
    o0 = wmma_bf16(pf, load_frag_bf16(vch + (0  + ml) * kVStride + c0), o0);
    o1 = wmma_bf16(pf, load_frag_bf16(vch + (16 + ml) * kVStride + c0), o1);
    o2 = wmma_bf16(pf, load_frag_bf16(vch + (32 + ml) * kVStride + c0), o2);
    o3 = wmma_bf16(pf, load_frag_bf16(vch + (48 + ml) * kVStride + c0), o3);

    __syncthreads();                  // done reading cur buffer
  }

  // ---- normalize and write ctx (bf16) in merged [B, S, H] layout ----
  const int h = bh % kNH;
#pragma unroll
  for (int r = 0; r < 8; ++r) {
    const float inv = 1.0f / lrun[r];
    const int   q   = q0 + r + 8 * lh;
    u16* crow = ctx + ((size_t)b * kS + q) * kH + h * kHD + ml;
    crow[0]  = f2bf_u(o0[r] * inv);
    crow[16] = f2bf_u(o1[r] * inv);
    crow[32] = f2bf_u(o2[r] * inv);
    crow[48] = f2bf_u(o3[r] * inv);
  }
}

// ---------------------------------------------------------------------------
// Kernel 3: output projection + bias + residual.  y = ctx @ Wo^T + bo + x
// ---------------------------------------------------------------------------
__global__ void __launch_bounds__(256)
out_proj_kernel(const u16* __restrict__ ctxb, const u16* __restrict__ Wob,
                const float* __restrict__ bo, const float* __restrict__ x,
                float* __restrict__ y) {
  const int lane = threadIdx.x & 31;
  const int wave = threadIdx.x >> 5;
  const int lh   = lane >> 4;
  const int ml   = lane & 15;
  const int c0   = lh * 8;
  const int m0   = (blockIdx.x * 8 + wave) * 16;
  const int n0   = blockIdx.y * 16;

  v8f acc = {};
  const u16* arow = ctxb + (size_t)(m0 + ml) * kH + c0;
  const u16* brow = Wob  + (size_t)(n0 + ml) * kH + c0;
#pragma unroll 4
  for (int k0 = 0; k0 < kH; k0 += 32)
    acc = wmma_bf16(load_frag_bf16(arow + k0), load_frag_bf16(brow + k0), acc);

  const int   n   = n0 + ml;
  const float bb_ = bo[n];
#pragma unroll
  for (int r = 0; r < 8; ++r) {
    const int    m   = m0 + r + 8 * lh;
    const size_t idx = (size_t)m * kH + n;
    y[idx] = acc[r] + bb_ + x[idx];
  }
}

// ---------------------------------------------------------------------------
// Kernel 4: LayerNorm over last dim (768).  One wave per row.
// ---------------------------------------------------------------------------
__global__ void __launch_bounds__(256)
layernorm_kernel(const float* __restrict__ y, const float* __restrict__ gamma,
                 const float* __restrict__ beta, float* __restrict__ out) {
  const int lane = threadIdx.x & 31;
  const int wave = threadIdx.x >> 5;
  const int row  = blockIdx.x * 8 + wave;
  const float* yr = y + (size_t)row * kH;

  float s = 0.0f, s2 = 0.0f;
  for (int i = lane; i < kH; i += 32) {
    const float v = yr[i];
    s += v; s2 += v * v;
  }
#pragma unroll
  for (int off = 1; off < 32; off <<= 1) {
    s  += __shfl_xor(s,  off, 32);
    s2 += __shfl_xor(s2, off, 32);
  }
  const float mean = s * (1.0f / kH);
  const float var  = s2 * (1.0f / kH) - mean * mean;
  const float rstd = rsqrtf(var + 1e-5f);

  float* orow = out + (size_t)row * kH;
  for (int i = lane; i < kH; i += 32)
    orow[i] = (yr[i] - mean) * rstd * gamma[i] + beta[i];
}

// ---------------------------------------------------------------------------
// Launch
// ---------------------------------------------------------------------------
extern "C" void kernel_launch(void* const* d_in, const int* in_sizes, int n_in,
                              void* d_out, int out_size, void* d_ws,
                              size_t ws_size, hipStream_t stream) {
  (void)in_sizes; (void)n_in; (void)out_size; (void)ws_size;

  const float* x     = (const float*)d_in[0];
  const float* amask = (const float*)d_in[1];
  const float* Wq    = (const float*)d_in[2];
  const float* bq    = (const float*)d_in[3];
  const float* Wk    = (const float*)d_in[4];
  const float* bk    = (const float*)d_in[5];
  const float* Wv    = (const float*)d_in[6];
  const float* bv    = (const float*)d_in[7];
  const float* Wo    = (const float*)d_in[8];
  const float* bo    = (const float*)d_in[9];
  const float* g     = (const float*)d_in[10];
  const float* be    = (const float*)d_in[11];
  float* out = (float*)d_out;

  // Workspace carve-up (all regions 16B-aligned by construction).
  const size_t NX = (size_t)kM * kH;          // 3,145,728 activation elems
  const size_t NW = (size_t)kH * kH;          // 589,824 weight elems
  char* w = (char*)d_ws;
  u16* XBF = (u16*)w;  w += NX * 2;
  u16* WQB = (u16*)w;  w += NW * 2;
  u16* WKB = (u16*)w;  w += NW * 2;
  u16* WVB = (u16*)w;  w += NW * 2;
  u16* WOB = (u16*)w;  w += NW * 2;
  u16* Qb  = (u16*)w;  w += NX * 2;           // [BH, S, HD]
  u16* Kb  = (u16*)w;  w += NX * 2;           // [BH, S, HD]
  u16* Vtb = (u16*)w;  w += NX * 2;           // [BH, HD, S] (transposed)
  u16* CTX = (u16*)w;  w += NX * 2;           // [B, S, H] bf16
  float* Y = (float*)w;                       // [B, S, H] f32

  // One-time f32 -> bf16 conversions.
  cvt_bf16_kernel<<<dim3((int)(NX / 2048)), 256, 0, stream>>>(x,  XBF, (int)NX);
  cvt_bf16_kernel<<<dim3((int)(NW / 2048)), 256, 0, stream>>>(Wq, WQB, (int)NW);
  cvt_bf16_kernel<<<dim3((int)(NW / 2048)), 256, 0, stream>>>(Wk, WKB, (int)NW);
  cvt_bf16_kernel<<<dim3((int)(NW / 2048)), 256, 0, stream>>>(Wv, WVB, (int)NW);
  cvt_bf16_kernel<<<dim3((int)(NW / 2048)), 256, 0, stream>>>(Wo, WOB, (int)NW);

  // QKV projections: 256 m-tiles x 48 n-tiles, 8 waves/block, z = Q/K/V.
  qkv_proj_kernel<<<dim3(kM / 16 / 8, kH / 16, 3), 256, 0, stream>>>(
      XBF, WQB, bq, WKB, bk, WVB, bv, Qb, Kb, Vtb);

  // Flash attention: 128 q-tiles (8 per block) x 24 (b,h) pairs.
  flash_attn_kernel<<<dim3(kS / 16 / 8, kBH), 256, 0, stream>>>(
      Qb, Kb, Vtb, amask, CTX);

  // Output projection + residual.
  out_proj_kernel<<<dim3(kM / 16 / 8, kH / 16), 256, 0, stream>>>(
      CTX, WOB, bo, x, Y);

  // LayerNorm -> d_out.
  layernorm_kernel<<<dim3(kM / 8), 256, 0, stream>>>(Y, g, be, out);
}
